// NasRNN_24455543783981
// MI455X (gfx1250) — matrix operand order repaired and checked by reading, not compiled, not run on hardware
//
#include <hip/hip_runtime.h>
#include <hip/hip_bf16.h>
#include <math.h>

// NAS-RNN fused persistent kernel for MI455X (gfx1250).
// SEQ=1000 steps of: s[g] = x_t @ W_ih[g] + m @ W_hh[g]  (8 gates, 256x256x256 each)
// then per-element NAS gate tree -> new c (kept in VGPRs), new m (bf16 ping-pong in ws).
//
// - bf16 WMMA (v_wmma_f32_16x16x32_bf16), fp32 accumulate, fp32 elementwise.
// - Per-WG weight slice (256 KB bf16, pre-packed into WMMA B-fragment layout)
//   is LDS-resident for the whole sequence (320 KB LDS/WGP).
// - Software-pipelined fragment loads: B(u+1) and A(next k-chunk) issued before WMMA(u)
//   so ds_load latency overlaps matrix ops instead of s_wait_dscnt 0 per WMMA.
// - m tile staged via gfx1250 async copy engine: global_load_async_to_lds_b128
//   (ASYNCcnt) + s_wait_asynccnt; x tile staged through VGPRs (needs f32->bf16).
// - Device-wide barrier per step via atomic counter (release/acquire), m double-buffered.
// - Grid = 64 WGs (8 batch-tiles x 8 hid-tiles) x 128 threads (4 wave32, 2x2 wave grid).

#define SEQ    1000
#define BATCH  256
#define IN_DIM 256
#define HID    256

typedef __attribute__((ext_vector_type(16))) __bf16 v16bf;
typedef __attribute__((ext_vector_type(8)))  float  v8f;

// Workspace layout (bytes):
//   [0,        2097152)  packed bf16 weights: 2(type) x 8(gate) x 16(n-tile) x 8(k-chunk)
//                        blocks of 32 lanes x 16 bf16 (WMMA B-fragment layout)
//   [2097152,  2228224)  m ping  (bf16, 256x256)
//   [2228224,  2359296)  m pong  (bf16, 256x256)
//   [2359296,  2359300)  barrier counter (u32)
#define WS_WPACK 0
#define WS_M0    2097152
#define WS_M1    2228224
#define WS_CTR   2359296

__device__ __forceinline__ unsigned short f2bf(float f) {
  unsigned u = __float_as_uint(f);
  u += 0x7FFFu + ((u >> 16) & 1u);   // round-to-nearest-even
  return (unsigned short)(u >> 16);
}
__device__ __forceinline__ float sigmoidf_(float x) {
  return 1.0f / (1.0f + __expf(-x));
}
__device__ __forceinline__ float tanhf_(float x) {
  float e = __expf(-2.0f * fabsf(x));
  float r = (1.0f - e) / (1.0f + e);
  return copysignf(r, x);
}

// 32-byte WMMA operand fragment (two b128 LDS beats)
struct Frag { uint4 a, b; };
__device__ __forceinline__ Frag ld2(const unsigned short* p, unsigned strideUshorts) {
  Frag f;
  f.a = *(const uint4*)p;
  f.b = *(const uint4*)(p + strideUshorts);
  return f;
}
__device__ __forceinline__ v16bf asV(const Frag& f) {
  union { Frag f; v16bf v; } u;
  u.f = f;
  return u.v;
}

// ---- init: pack weights fp32 -> bf16 in per-lane B-fragment layout, init m=1, ctr=0
__global__ void nas_init(const float* __restrict__ w_ih,
                         const float* __restrict__ w_hh,
                         unsigned short* __restrict__ wpack,
                         unsigned short* __restrict__ m0,
                         unsigned int* __restrict__ ctr) {
  unsigned p = blockIdx.x * blockDim.x + threadIdx.x;
  if (p < 1048576u) {
    unsigned i    = p & 15u;          // element within lane (ascending K)
    unsigned lane = (p >> 4) & 31u;   // lane
    unsigned kc   = (p >> 9) & 7u;    // K chunk of 32
    unsigned nt   = (p >> 12) & 15u;  // n-tile of 16
    unsigned g    = (p >> 16) & 7u;   // gate
    unsigned ts   = (p >> 19) & 1u;   // 0=W_ih, 1=W_hh
    // B 32x16 bf16 layout: lanes 0-15 hold K=0..7/16..23, lanes 16-31 hold K=8..15/24..31; N=lane%16
    unsigned k = kc * 32u + ((lane & 16u) ? 8u : 0u) + (i < 8u ? i : i + 8u);
    unsigned n = nt * 16u + (lane & 15u);
    const float* w = ts ? w_hh : w_ih;
    wpack[p] = f2bf(w[(g * 256u + k) * 256u + n]);
  }
  if (p < 65536u) m0[p] = 0x3F80u;    // bf16(1.0)
  if (p == 0) *ctr = 0u;
}

__global__ __launch_bounds__(128, 1)
void nas_rnn_persistent(const float* __restrict__ x,              // [SEQ,BATCH,IN]
                        const unsigned short* __restrict__ wpack, // packed bf16 weights
                        unsigned short* __restrict__ mbuf0,
                        unsigned short* __restrict__ mbuf1,
                        unsigned int* __restrict__ ctr,
                        float* __restrict__ out,                  // [BATCH,HID]
                        unsigned nwg) {
  extern __shared__ unsigned short lds[];
  unsigned short* wlds = lds;                  // 131072 ushort = 256 KB weight slice
  unsigned short* xA   = lds + 131072;         // 32 x 264 (padded) bf16 x tile
  unsigned short* mA   = lds + 131072 + 8448;  // 32 x 264 (padded) bf16 m tile

  const unsigned tid  = threadIdx.x;
  const unsigned lane = tid & 31u;
  const unsigned wave = tid >> 5;
  const unsigned wb   = wave >> 1;        // wave batch sub-tile (0..1)
  const unsigned hp   = wave & 1u;        // wave hid  sub-tile (0..1)
  const unsigned bt   = blockIdx.x >> 3;  // batch tile 0..7 (32 rows)
  const unsigned ht   = blockIdx.x & 7u;  // hid tile   0..7 (32 cols)

  // ---- one-time: stage this WG's 256 KB weight slice into LDS
  {
    const uint4* src = (const uint4*)wpack;
    uint4* dst = (uint4*)wlds;
    for (unsigned idx = tid; idx < 16384u; idx += 128u) {
      unsigned bl = idx >> 6;        // LDS block (1 KB = 64 uint4)
      unsigned w  = idx & 63u;
      unsigned kc = bl & 7u;
      unsigned np = (bl >> 3) & 1u;
      unsigned g  = (bl >> 4) & 7u;
      unsigned ts = bl >> 7;
      unsigned gb = (((ts * 8u + g) * 16u + (ht * 2u + np)) * 8u + kc);
      dst[idx] = src[gb * 64u + w];
    }
  }

  // persistent cell state c lives in VGPRs for all 1000 steps
  v8f cacc;
#pragma unroll
  for (int i = 0; i < 8; ++i) cacc[i] = 1.0f;

  // D-matrix 16x16 f32 layout: VGPR i, lanes0-15: M=i,N=lane; lanes16-31: M=i+8,N=lane-16
  const unsigned colN  = ht * 32u + hp * 16u + (lane & 15u);
  const unsigned rowB  = bt * 32u + wb * 16u + ((lane & 16u) ? 8u : 0u);
  // A-matrix 16x32 bf16: lane row = lane%16; lane<16: K 0..7/16..23, lane>=16: K 8..15/24..31
  const unsigned ar    = wb * 16u + (lane & 15u);
  const unsigned acsel = (lane & 16u) ? 8u : 0u;

  // fragment address helpers: flattened WMMA index u = ts*64 + kc*8 + g, block = u>>3
  auto aAddr = [&](int blk) -> const unsigned short* {
    const unsigned short* A = (blk & 8) ? mA : xA;
    return A + ar * 264u + (unsigned)(blk & 7) * 32u + acsel;
  };
  auto bAddr = [&](int u) -> const unsigned short* {
    unsigned ts2 = (unsigned)u >> 6, kc = ((unsigned)u >> 3) & 7u, g = (unsigned)u & 7u;
    return wlds + ((((ts2 * 8u + g) * 2u + hp) * 8u + kc) * 512u) + lane * 16u;
  };

  const unsigned mLdsBase = (unsigned)(uintptr_t)mA;  // generic LDS ptr low 32 = LDS byte addr

  for (int t = 0; t < SEQ; ++t) {
    __syncthreads();  // protect LDS staging buffers from previous step's readers

    // ---- stage m tile (32 x 256 bf16) via gfx1250 async copy engine (ASYNCcnt)
    {
      const unsigned short* mread = (t & 1) ? mbuf1 : mbuf0;
      const unsigned short* gbase = mread + (size_t)bt * 32u * HID;
#pragma unroll
      for (int j = 0; j < 8; ++j) {
        unsigned q  = (unsigned)j * 128u + tid;   // 0..1023 16-byte chunks
        unsigned r  = q >> 5, c8 = q & 31u;       // 32 rows x 32 chunks/row
        unsigned long long ga = (unsigned long long)(uintptr_t)(gbase + r * 256u + c8 * 8u);
        unsigned la = mLdsBase + r * 528u + c8 * 16u;
        asm volatile("global_load_async_to_lds_b128 %0, %1, off"
                     :: "v"(la), "v"(ga) : "memory");
      }
    }

    // ---- stage x_t tile (32 x 256 fp32 -> bf16, padded rows) through VGPRs
    {
      const float4* xs = (const float4*)(x + ((size_t)t * BATCH + bt * 32u) * IN_DIM);
#pragma unroll
      for (int j = 0; j < 16; ++j) {
        unsigned q = (unsigned)j * 128u + tid;   // 0..2047 float4
        unsigned r = q >> 6, c4 = q & 63u;
        float4 v = xs[r * 64u + c4];
        ushort4 b;
        b.x = f2bf(v.x); b.y = f2bf(v.y); b.z = f2bf(v.z); b.w = f2bf(v.w);
        *(ushort4*)&xA[r * 264u + c4 * 4u] = b;
      }
      if (t + 1 < SEQ)  // global_prefetch next step's x tile
        __builtin_prefetch(x + ((size_t)(t + 1) * BATCH + bt * 32u) * IN_DIM + tid * 64u, 0, 0);
    }

    asm volatile("s_wait_asynccnt 0" ::: "memory");  // m tile resident in LDS
    __syncthreads();

    // ---- s[g] = x W_ih[g] + m W_hh[g]
    // 128 WMMAs per wave, software pipelined: B(u+1) & A(next blk) load before WMMA(u)
    v8f acc[8] = {};
    Frag a_cur = ld2(aAddr(0), 16);
    Frag b_cur = ld2(bAddr(0), 8);
    Frag a_nxt = a_cur;
#pragma unroll
    for (int u = 0; u < 128; ++u) {
      Frag b_nxt = ld2(bAddr((u + 1) & 127), 8);
      if ((u & 7) == 6) a_nxt = ld2(aAddr(((u >> 3) + 1) & 15), 16);
      acc[u & 7] = __builtin_amdgcn_wmma_f32_16x16x32_bf16(
          false, asV(a_cur), false, asV(b_cur), (short)0, acc[u & 7], false, false);
      b_cur = b_nxt;
      if ((u & 7) == 7) a_cur = a_nxt;
    }

    // ---- NAS gate tree (per-lane, fp32) + state update
    unsigned short* mwrite = (t & 1) ? mbuf0 : mbuf1;
#pragma unroll
    for (int i = 0; i < 8; ++i) {
      float l10 = sigmoidf_(acc[0][i]);
      float l11 = fmaxf(acc[1][i], 0.0f);
      float l12 = sigmoidf_(acc[2][i]);
      float l13 = fmaxf(acc[3][i], 0.0f);
      float l14 = tanhf_(acc[4][i]);
      float l15 = sigmoidf_(acc[5][i]);
      float l16 = tanhf_(acc[6][i]);
      float l17 = sigmoidf_(acc[7][i]);
      float l20 = tanhf_(l10 * l11);
      float l21 = tanhf_(l12 + l13);
      float l22 = tanhf_(l14 * l15);
      float l23 = sigmoidf_(l16 + l17);
      float l20v2 = tanhf_(l20 + cacc[i]);
      float nc  = l20v2 * l21;
      float l31 = tanhf_(l22 + l23);
      float nm  = tanhf_(nc * l31);
      cacc[i] = nc;
      unsigned idx = (rowB + (unsigned)i) * HID + colN;
      mwrite[idx] = f2bf(nm);
      if (t == SEQ - 1) out[idx] = nm;
    }

    // ---- device-wide barrier: release this step's m, wait for all 64 WGs
    __threadfence();
    __syncthreads();
    if (tid == 0) {
      __hip_atomic_fetch_add(ctr, 1u, __ATOMIC_RELEASE, __HIP_MEMORY_SCOPE_AGENT);
      const unsigned target = (unsigned)(t + 1) * nwg;
      while (__hip_atomic_load(ctr, __ATOMIC_ACQUIRE, __HIP_MEMORY_SCOPE_AGENT) < target)
        __builtin_amdgcn_s_sleep(1);
    }
    __syncthreads();
  }
}

extern "C" void kernel_launch(void* const* d_in, const int* in_sizes, int n_in,
                              void* d_out, int out_size, void* d_ws, size_t ws_size,
                              hipStream_t stream) {
  (void)in_sizes; (void)n_in; (void)out_size; (void)ws_size;
  const float* x    = (const float*)d_in[0];
  const float* w_ih = (const float*)d_in[1];
  const float* w_hh = (const float*)d_in[2];

  unsigned short* wpack = (unsigned short*)((char*)d_ws + WS_WPACK);
  unsigned short* m0    = (unsigned short*)((char*)d_ws + WS_M0);
  unsigned short* m1    = (unsigned short*)((char*)d_ws + WS_M1);
  unsigned int*   ctr   = (unsigned int*)((char*)d_ws + WS_CTR);

  nas_init<<<4096, 256, 0, stream>>>(w_ih, w_hh, wpack, m0, ctr);

  const unsigned nwg = 64;  // 8 batch-tiles x 8 hid-tiles
  const size_t shmem = (size_t)(131072 + 2 * 8448) * sizeof(unsigned short);  // 295,936 B
  nas_rnn_persistent<<<dim3(nwg), dim3(128), shmem, stream>>>(
      x, wpack, m0, m1, ctr, (float*)d_out, nwg);
}
